// nEq_NLMP_17368847745646
// MI455X (gfx1250) — compile-verified, matrix operand order checked by reading, not run on hardware
//
#include <hip/hip_runtime.h>
#include <hip/hip_bf16.h>
#include <math.h>

typedef __bf16 bf16_t;
typedef __attribute__((ext_vector_type(16))) __bf16 v16bf;
typedef __attribute__((ext_vector_type(8)))  float  v8f;
typedef __attribute__((ext_vector_type(4)))  unsigned int v4u;

constexpr int Nn = 50000, Ee = 800000, Dd = 32, NFEc = 16;

// bf16 element offsets inside d_ws (all byte offsets 16B aligned)
constexpr int OFF_EV1 = 0;      // ev_w1^T  [128][96]
constexpr int OFF_EV2 = 12288;  // ev_w2^T  [16][128]
constexpr int OFF_FC1 = 14336;  // fc_w1^T  [64][32]  (K padded 22->32 with zeros)
constexpr int OFF_FC2 = 16384;  // fc_w2^T  [512][64]
constexpr int OFF_NU1 = 49152;  // nu_w1^T  [128][64]
constexpr int OFF_NU2 = 57344;  // nu_w2^T  [32][128]

// ---------------- prep kernels ----------------
__global__ void zerof(float* p, int n) {
  int i = blockIdx.x * blockDim.x + threadIdx.x;
  if (i < n) p[i] = 0.f;
}

// dst[n][k] = (bf16) src[k][n], zero-padded in K
__global__ void cvtT(const float* __restrict__ src, bf16_t* __restrict__ dst,
                     int K, int N, int Kpad) {
  int i = blockIdx.x * blockDim.x + threadIdx.x;
  if (i >= N * Kpad) return;
  int n = i / Kpad, k = i - n * Kpad;
  float v = (k < K) ? src[k * N + n] : 0.f;
  dst[n * Kpad + k] = (bf16_t)v;
}

// ---------------- WMMA helpers ----------------
// Load a 16x32 bf16 fragment (A layout; B uses the same layout on the [N][K]
// transposed weight arrays) from a row-major LDS array, width ldk elems.
__device__ __forceinline__ v16bf frag16(const bf16_t* base, int ldk, int row0, int kb) {
  int lane = threadIdx.x & 31;
  int r  = row0 + (lane & 15);
  int k0 = kb + ((lane >> 4) << 3);  // half 0: K+0..7,+16..23 ; half 1: +8..15,+24..31
  const bf16_t* p = base + r * ldk + k0;
  union { v16bf v; v4u u[2]; } t;
  t.u[0] = *(const v4u*)(p);
  t.u[1] = *(const v4u*)(p + 16);
  return t.v;
}

__device__ __forceinline__ v8f wmma_bf16(v16bf a, v16bf b, v8f c) {
  return __builtin_amdgcn_wmma_f32_16x16x32_bf16(false, a, false, b, (short)0, c,
                                                 false, false);
}

__device__ __forceinline__ float relu(float f) { return __builtin_fmaxf(f, 0.f); }

// ---------------- edge kernel ----------------
// dynamic LDS layout (bytes):
//   [0,24576)        ev_w1^T
//   [24576,28672)    ev_w2^T
//   [28672,32768)    fc_w1^T
//   [32768,98304)    fc_w2^T
//   per-wave region @ 98304 + waveId*11520:
//     x1 [16][96] bf16 (3072) | x2 [16][32] bf16 (1024) | h1 [16][128] bf16 (4096)
//     h2 [16][64] bf16 (2048) | vv [16][16] f32 (1024)  | nr [16] f32 (64)
//     si [16] int (64) | di [16] int (64)
__global__ void __launch_bounds__(128, 1)
edge_kernel(const float* __restrict__ hn, const float* __restrict__ he,
            const float* __restrict__ pos, const float* __restrict__ fes,
            const float* __restrict__ norm, const int* __restrict__ esrc,
            const int* __restrict__ edst, const bf16_t* __restrict__ ws,
            float* __restrict__ out_hen, float* __restrict__ nodeacc) {
  extern __shared__ char smem[];
  bf16_t* w_ev1 = (bf16_t*)(smem);
  bf16_t* w_ev2 = (bf16_t*)(smem + 24576);
  bf16_t* w_fc1 = (bf16_t*)(smem + 28672);
  bf16_t* w_fc2 = (bf16_t*)(smem + 32768);

  const int tid = threadIdx.x;
  {  // stage the 4 edge-side weight matrices (49152 bf16 = 24576 dwords)
    const unsigned int* s = (const unsigned int*)ws;
    unsigned int* d = (unsigned int*)smem;
    for (int i = tid; i < 24576; i += blockDim.x) d[i] = s[i];
  }
  __syncthreads();

  const int lane = tid & 31;
  const int waveId = tid >> 5;
  char* pw = smem + 98304 + waveId * 11520;
  bf16_t* x1 = (bf16_t*)(pw);
  bf16_t* x2 = (bf16_t*)(pw + 3072);
  bf16_t* h1 = (bf16_t*)(pw + 4096);
  bf16_t* h2 = (bf16_t*)(pw + 8192);
  float*  vv = (float*)(pw + 10240);
  float*  nr = (float*)(pw + 11264);
  int*    si = (int*)(pw + 11328);
  int*    di = (int*)(pw + 11392);

  const int numTiles = Ee / 16;
  const int gw = blockIdx.x * (blockDim.x >> 5) + waveId;
  const int stride = gridDim.x * (blockDim.x >> 5);
  const int col = lane & 15;
  const int rbase = (lane >> 4) << 3;  // rows 0..7 or 8..15 of the D fragment

  for (int t = gw; t < numTiles; t += stride) {
    const int e0 = t * 16;
    if (lane < 16) {
      si[lane] = esrc[e0 + lane];
      di[lane] = edst[e0 + lane];
      nr[lane] = norm[e0 + lane];
    }
    // gather inputs -> bf16 LDS tiles (intra-wave DS ops are in order; no barrier)
    #pragma unroll 4
    for (int r = 0; r < 16; r++) {
      int s = si[r], d2 = di[r];
      int e = e0 + r;
      x1[r * 96 + lane]      = (bf16_t)he[e * Dd + lane];
      x1[r * 96 + 32 + lane] = (bf16_t)hn[s * Dd + lane];
      x1[r * 96 + 64 + lane] = (bf16_t)hn[d2 * Dd + lane];
      float xv = 0.f;
      if (lane < 3)       xv = pos[s * 3 + lane];
      else if (lane < 6)  xv = pos[d2 * 3 + lane - 3];
      else if (lane < 22) xv = fes[e * NFEc + (lane - 6)];
      x2[r * 32 + lane] = (bf16_t)xv;
    }

    // prime the HBM pipe for the next tile while this tile's WMMAs run
    {
      int tn = t + stride;
      if (tn < numTiles && lane < 16) {
        __builtin_prefetch(&he[(size_t)(tn * 16 + lane) * Dd], 0, 3);
        __builtin_prefetch(&fes[(size_t)(tn * 16 + lane) * NFEc], 0, 3);
      }
    }

    // ---- MLP1 layer1: [16x96] @ [96x128] -> relu -> h1 ----
    v16bf a0 = frag16(x1, 96, 0, 0);
    v16bf a1 = frag16(x1, 96, 0, 32);
    v16bf a2 = frag16(x1, 96, 0, 64);
    #pragma unroll
    for (int nt = 0; nt < 8; nt++) {
      v8f acc = {};
      acc = wmma_bf16(a0, frag16(w_ev1, 96, nt * 16, 0), acc);
      acc = wmma_bf16(a1, frag16(w_ev1, 96, nt * 16, 32), acc);
      acc = wmma_bf16(a2, frag16(w_ev1, 96, nt * 16, 64), acc);
      #pragma unroll
      for (int r = 0; r < 8; r++)
        h1[(rbase + r) * 128 + nt * 16 + col] = (bf16_t)relu(acc[r]);
    }
    // ---- MLP1 layer2: [16x128] @ [128x16] -> v (f32 in LDS) ----
    {
      v8f acc = {};
      #pragma unroll
      for (int kc = 0; kc < 4; kc++)
        acc = wmma_bf16(frag16(h1, 128, 0, kc * 32),
                        frag16(w_ev2, 128, 0, kc * 32), acc);
      #pragma unroll
      for (int r = 0; r < 8; r++) vv[(rbase + r) * 16 + col] = acc[r];
    }
    // ---- MLP2 layer1: [16x32(pad)] @ [32x64] -> relu -> h2 ----
    {
      v16bf ax = frag16(x2, 32, 0, 0);
      #pragma unroll
      for (int nt = 0; nt < 4; nt++) {
        v8f acc = {};
        acc = wmma_bf16(ax, frag16(w_fc1, 32, nt * 16, 0), acc);
        #pragma unroll
        for (int r = 0; r < 8; r++)
          h2[(rbase + r) * 64 + nt * 16 + col] = (bf16_t)relu(acc[r]);
      }
    }
    // ---- MLP2 layer2 fused with tp contraction ----
    // W col tile nt covers flat cols [16nt,16nt+16) = (v = nt/2, o-half = nt&1),
    // so tp accumulates each D fragment row-scaled by v[e, nt/2].
    v8f tp0 = {}, tp1 = {};
    v16bf ha0 = frag16(h2, 64, 0, 0);
    v16bf ha1 = frag16(h2, 64, 0, 32);
    #pragma unroll
    for (int vi = 0; vi < 16; vi++) {
      float s[8];
      #pragma unroll
      for (int r = 0; r < 8; r++) s[r] = vv[(rbase + r) * 16 + vi];
      {
        int nt = vi * 2;
        v8f acc = {};
        acc = wmma_bf16(ha0, frag16(w_fc2, 64, nt * 16, 0), acc);
        acc = wmma_bf16(ha1, frag16(w_fc2, 64, nt * 16, 32), acc);
        #pragma unroll
        for (int r = 0; r < 8; r++) tp0[r] += s[r] * acc[r];
      }
      {
        int nt = vi * 2 + 1;
        v8f acc = {};
        acc = wmma_bf16(ha0, frag16(w_fc2, 64, nt * 16, 0), acc);
        acc = wmma_bf16(ha1, frag16(w_fc2, 64, nt * 16, 32), acc);
        #pragma unroll
        for (int r = 0; r < 8; r++) tp1[r] += s[r] * acc[r];
      }
    }
    // ---- hen = he + tp; scatter hen*norm into node accumulator ----
    #pragma unroll
    for (int r = 0; r < 8; r++) {
      int row = rbase + r;
      int e = e0 + row;
      float nw = nr[row];
      int dn = di[row];
      float o0 = he[e * Dd + col] + tp0[r];
      float o1 = he[e * Dd + 16 + col] + tp1[r];
      out_hen[e * Dd + col] = o0;
      out_hen[e * Dd + 16 + col] = o1;
      atomicAdd(&nodeacc[dn * Dd + col], o0 * nw);
      atomicAdd(&nodeacc[dn * Dd + 16 + col], o1 * nw);
    }
  }
}

// ---------------- node kernel ----------------
__global__ void __launch_bounds__(128, 1)
node_kernel(const float* __restrict__ hn, const bf16_t* __restrict__ ws,
            float* __restrict__ outp /* hnn region == node accumulator */) {
  __shared__ __align__(16) bf16_t w1T[128 * 64];
  __shared__ __align__(16) bf16_t w2T[32 * 128];
  __shared__ __align__(16) bf16_t x3s[4][16 * 64];
  __shared__ __align__(16) bf16_t hs[4][16 * 128];

  const int tid = threadIdx.x;
  {
    const unsigned int* s1 = (const unsigned int*)(ws + OFF_NU1);
    unsigned int* d1 = (unsigned int*)w1T;
    for (int i = tid; i < 4096; i += blockDim.x) d1[i] = s1[i];
    const unsigned int* s2 = (const unsigned int*)(ws + OFF_NU2);
    unsigned int* d2 = (unsigned int*)w2T;
    for (int i = tid; i < 2048; i += blockDim.x) d2[i] = s2[i];
  }
  __syncthreads();

  const int lane = tid & 31, waveId = tid >> 5;
  const int tile = blockIdx.x * 4 + waveId;
  if (tile >= Nn / 16) return;

  bf16_t* x3 = x3s[waveId];
  bf16_t* hh = hs[waveId];
  const int n0 = tile * 16;
  const int col = lane & 15, rbase = (lane >> 4) << 3;

  for (int r = 0; r < 16; r++) {
    int n = n0 + r;
    x3[r * 64 + lane]      = (bf16_t)hn[n * Dd + lane];
    x3[r * 64 + 32 + lane] = (bf16_t)outp[n * Dd + lane];  // aggregated messages
  }
  v16bf a0 = frag16(x3, 64, 0, 0);
  v16bf a1 = frag16(x3, 64, 0, 32);
  #pragma unroll
  for (int nt = 0; nt < 8; nt++) {
    v8f acc = {};
    acc = wmma_bf16(a0, frag16(w1T, 64, nt * 16, 0), acc);
    acc = wmma_bf16(a1, frag16(w1T, 64, nt * 16, 32), acc);
    #pragma unroll
    for (int r = 0; r < 8; r++)
      hh[(rbase + r) * 128 + nt * 16 + col] = (bf16_t)relu(acc[r]);
  }
  v16bf b0 = frag16(hh, 128, 0, 0);
  v16bf b1 = frag16(hh, 128, 0, 32);
  v16bf b2 = frag16(hh, 128, 0, 64);
  v16bf b3 = frag16(hh, 128, 0, 96);
  #pragma unroll
  for (int nt = 0; nt < 2; nt++) {
    v8f acc = {};
    acc = wmma_bf16(b0, frag16(w2T, 128, nt * 16, 0), acc);
    acc = wmma_bf16(b1, frag16(w2T, 128, nt * 16, 32), acc);
    acc = wmma_bf16(b2, frag16(w2T, 128, nt * 16, 64), acc);
    acc = wmma_bf16(b3, frag16(w2T, 128, nt * 16, 96), acc);
    #pragma unroll
    for (int r = 0; r < 8; r++) {
      int n = n0 + rbase + r;
      int o = nt * 16 + col;
      outp[n * Dd + o] = hn[n * Dd + o] + acc[r];  // residual, overwrites accumulator
    }
  }
}

// ---------------- launch ----------------
extern "C" void kernel_launch(void* const* d_in, const int* in_sizes, int n_in,
                              void* d_out, int out_size, void* d_ws, size_t ws_size,
                              hipStream_t stream) {
  const float* hn    = (const float*)d_in[0];
  const float* he    = (const float*)d_in[1];
  const float* pos   = (const float*)d_in[2];
  const float* fes   = (const float*)d_in[3];
  const float* norm  = (const float*)d_in[4];
  const int*   esrc  = (const int*)d_in[5];
  const int*   edst  = (const int*)d_in[6];
  const float* ev_w1 = (const float*)d_in[7];
  const float* ev_w2 = (const float*)d_in[8];
  const float* fc_w1 = (const float*)d_in[9];
  const float* fc_w2 = (const float*)d_in[10];
  const float* nu_w1 = (const float*)d_in[11];
  const float* nu_w2 = (const float*)d_in[12];
  float* out = (float*)d_out;  // [0,N*D): hnn (used first as node accumulator)
                               // [N*D, N*D+E*D): hen
  bf16_t* wsb = (bf16_t*)d_ws;

  zerof<<<(Nn * Dd + 255) / 256, 256, 0, stream>>>(out, Nn * Dd);
  cvtT<<<(12288 + 255) / 256, 256, 0, stream>>>(ev_w1, wsb + OFF_EV1, 96, 128, 96);
  cvtT<<<(2048 + 255) / 256, 256, 0, stream>>>(ev_w2, wsb + OFF_EV2, 128, 16, 128);
  cvtT<<<(2048 + 255) / 256, 256, 0, stream>>>(fc_w1, wsb + OFF_FC1, 22, 64, 32);
  cvtT<<<(32768 + 255) / 256, 256, 0, stream>>>(fc_w2, wsb + OFF_FC2, 64, 512, 64);
  cvtT<<<(8192 + 255) / 256, 256, 0, stream>>>(nu_w1, wsb + OFF_NU1, 64, 128, 64);
  cvtT<<<(4096 + 255) / 256, 256, 0, stream>>>(nu_w2, wsb + OFF_NU2, 128, 32, 128);

  size_t shmem = 98304 + 4 * 11520;  // weights + 4 per-wave scratch regions
  edge_kernel<<<1250, 128, shmem, stream>>>(hn, he, pos, fes, norm, esrc, edst, wsb,
                                            out + (size_t)Nn * Dd, out);
  node_kernel<<<(3125 + 3) / 4, 128, 0, stream>>>(hn, wsb, out);
}
